// Mamba_Cnn_30949534335444
// MI455X (gfx1250) — matrix operand order repaired and checked by reading, not compiled, not run on hardware
//
// MI455X (gfx1250, CDNA5) implementation of the dual-encoder U-Net + attention head.
// f16 operands + v_wmma_f32_16x16x32_f16 implicit-GEMM for every conv/GEMM, f32 accum,
// fused bias epilogue, in-place BN+activation. 64x128 output tiles per 256-thread
// (8 x wave32) block, double-buffered LDS. NEW this round: Tensor Data Mover
// (tensor_load_to_lds + s_wait_tensorcnt) feeds the A tile (conv weights) and both
// A/B tiles in the plain GEMM, replacing VALU-addressed loads + ds_store.
#include <hip/hip_runtime.h>
#include <math.h>

typedef __attribute__((ext_vector_type(16))) _Float16 v16h;
typedef __attribute__((ext_vector_type(8)))  float    v8f;
typedef unsigned int u32x4 __attribute__((ext_vector_type(4)));
typedef int          i32x8 __attribute__((ext_vector_type(8)));
typedef int          i32x4 __attribute__((ext_vector_type(4)));
typedef _Float16 half_t;

#define DEVI __device__ __forceinline__
#define HAVE_TDM __has_builtin(__builtin_amdgcn_tensor_load_to_lds)

DEVI float gelu_f(float x){ return 0.5f*x*(1.f+erff(x*0.70710678118654752f)); }
DEVI float sigm_f(float x){ return 1.f/(1.f+__expf(-x)); }
DEVI float act_f(float x,int a){
  if(a==1) return gelu_f(x);
  if(a==2) return fmaxf(x,0.f);
  if(a==3) return sigm_f(x);
  return x;
}

#if HAVE_TDM
// One rectangular f16 tile (tileX elems wide, tileY rows) from a row-major tensor
// (row stride = strideElems) into LDS at byte address `lds`. dim0/dim1 are the
// REMAINING tensor extents measured from the tile origin -> TDM zero-fills OOB.
DEVI void tdm_load_2d(unsigned lds,const void* g,unsigned tileX,unsigned tileY,
                      unsigned dim0,unsigned dim1,unsigned strideElems){
  unsigned long long ga=(unsigned long long)g;
  u32x4 g0={ 1u,                                   // count=1, user descriptor
             lds,                                  // lds_addr [63:32]
             (unsigned)ga,                         // global_addr [95:64]
             (unsigned)((ga>>32)&0x01ffffffu)|(2u<<30) }; // global_addr hi | type=2
  i32x8 g1={ (int)(1u<<16),                        // data_size=1 (2 bytes/elem)
             (int)((dim0&0xffffu)<<16),            // tensor_dim0 [15:0]
             (int)((dim0>>16)|((dim1&0xffffu)<<16)),// dim0 hi | tensor_dim1 lo
             (int)((dim1>>16)|(tileX<<16)),        // dim1 hi | tile_dim0
             (int)(tileY&0xffffu),                 // tile_dim1 (tile_dim2=0)
             (int)strideElems,                     // tensor_dim0_stride [31:0]
             0, 0 };
  i32x4 z4={0,0,0,0};
#if defined(__clang_major__) && (__clang_major__>=23)
  i32x8 z8={0,0,0,0,0,0,0,0};
  __builtin_amdgcn_tensor_load_to_lds(g0,g1,z4,z4,z8,0);
#else
  __builtin_amdgcn_tensor_load_to_lds(g0,g1,z4,z4,0);
#endif
}
DEVI unsigned lds_off(const void* p){ return (unsigned)(unsigned long long)p; } // ISA 10.2: LDS byte offset = addr[31:0]
#endif

// ---------------------------------------------------------------- elementwise
__global__ void k_f16(const float* __restrict__ s, half_t* __restrict__ d, long n){
  long i=(long)blockIdx.x*blockDim.x+threadIdx.x;
  if(i<n) d[i]=(half_t)s[i];
}
__global__ void k_packW(const float* __restrict__ s, half_t* __restrict__ d,int R,int K,int Kpad){
  long i=(long)blockIdx.x*blockDim.x+threadIdx.x;
  long n=(long)R*Kpad;
  if(i<n){ int r=(int)(i/Kpad), k=(int)(i%Kpad);
    d[i]=(k<K)?(half_t)s[(long)r*K+k]:(half_t)0.f; }
}
__global__ void k_perm021(const float* __restrict__ s,float* __restrict__ d,int B,int X,int Y){
  long i=(long)blockIdx.x*blockDim.x+threadIdx.x; long n=(long)B*X*Y;
  if(i<n){ int b=(int)(i/((long)X*Y)); long r=i%((long)X*Y); int x=(int)(r/Y), y=(int)(r%Y);
    d[((long)b*Y+y)*X+x]=s[i]; }
}
__global__ void k_perm021h(const float* __restrict__ s,half_t* __restrict__ d,int B,int X,int Y){
  long i=(long)blockIdx.x*blockDim.x+threadIdx.x; long n=(long)B*X*Y;
  if(i<n){ int b=(int)(i/((long)X*Y)); long r=i%((long)X*Y); int x=(int)(r/Y), y=(int)(r%Y);
    d[((long)b*Y+y)*X+x]=(half_t)s[i]; }
}
__global__ void k_chcpy(const float* __restrict__ s,float* __restrict__ d,
                        int B,int Cs,int c0,int Cn,int HW,int Cd,int d0){
  long i=(long)blockIdx.x*blockDim.x+threadIdx.x; long n=(long)B*Cn*HW;
  if(i>=n) return;
  int p=(int)(i%HW); long r=i/HW; int c=(int)(r%Cn); int b=(int)(r/Cn);
  d[((long)b*Cd+d0+c)*HW+p]=s[((long)b*Cs+c0+c)*HW+p];
}
__global__ void k_add(const float* __restrict__ a,const float* __restrict__ b,float* __restrict__ o,long n){
  long i=(long)blockIdx.x*blockDim.x+threadIdx.x; if(i<n) o[i]=a[i]+b[i];
}
__global__ void k_mul(const float* __restrict__ a,const float* __restrict__ b,float* __restrict__ o,long n){
  long i=(long)blockIdx.x*blockDim.x+threadIdx.x; if(i<n) o[i]=a[i]*b[i];
}
__global__ void k_unary(float* __restrict__ X,long n,int a){
  long i=(long)blockIdx.x*blockDim.x+threadIdx.x; if(i<n) X[i]=act_f(X[i],a);
}

// ---------------------------------------------------------------- batchnorm (train-mode batch stats)
__global__ __launch_bounds__(256) void k_bn_stats(const float* __restrict__ X,float2* __restrict__ mv,
                                                  int Bn,int C,int HW){
  int c=blockIdx.x, t=threadIdx.x;
  float s=0.f,s2=0.f; long cnt=(long)Bn*HW;
  for(long i=t;i<cnt;i+=256){
    long b=i/HW, p=i%HW;
    float v=X[((long)b*C+c)*HW+p];
    s+=v; s2+=v*v;
  }
  __shared__ float r1[256], r2[256];
  r1[t]=s; r2[t]=s2; __syncthreads();
  for(int st=128;st>0;st>>=1){ if(t<st){ r1[t]+=r1[t+st]; r2[t]+=r2[t+st]; } __syncthreads(); }
  if(t==0){ float m=r1[0]/(float)cnt; mv[c]=make_float2(m, r2[0]/(float)cnt - m*m); }
}
__global__ void k_bn_act(float* __restrict__ X,const float2* __restrict__ mv,
                         const float* __restrict__ g,const float* __restrict__ bt,
                         int C,int HW,long n,int act){
  long i=(long)blockIdx.x*blockDim.x+threadIdx.x;
  if(i>=n) return;
  int c=(int)((i/HW)%C);
  float2 m=mv[c];
  float v=(X[i]-m.x)*rsqrtf(m.y+1e-5f)*g[c]+bt[c];
  X[i]=act_f(v,act);
}

// ---------------------------------------------------------------- pool / resize / sobel / softmax
__global__ void k_maxpool2(const float* __restrict__ s,float* __restrict__ d,int B,int C,int H,int W){
  int Ho=H>>1, Wo=W>>1; long n=(long)B*C*Ho*Wo;
  long i=(long)blockIdx.x*blockDim.x+threadIdx.x; if(i>=n) return;
  int wo=(int)(i%Wo); long r=i/Wo; int ho=(int)(r%Ho); r/=Ho; int c=(int)(r%C); int b=(int)(r/C);
  const float* p=s+(((long)b*C+c)*H+ho*2)*W+wo*2;
  d[i]=fmaxf(fmaxf(p[0],p[1]),fmaxf(p[W],p[W+1]));
}
__global__ void k_resize(const float* __restrict__ s,float* __restrict__ d,
                         int B,int C,int Hs,int Ws,int Hd,int Wd){
  long n=(long)B*C*Hd*Wd;
  long i=(long)blockIdx.x*blockDim.x+threadIdx.x; if(i>=n) return;
  int wd=(int)(i%Wd); long r=i/Wd; int hd=(int)(r%Hd); r/=Hd; int c=(int)(r%C); int b=(int)(r/C);
  float sy=((hd+0.5f)*Hs)/Hd-0.5f, sx=((wd+0.5f)*Ws)/Wd-0.5f;
  int y0=(int)floorf(sy), x0=(int)floorf(sx);
  float fy=sy-(float)y0, fx=sx-(float)x0;
  int y1=y0+1, x1=x0+1;
  y0=y0<0?0:(y0>Hs-1?Hs-1:y0); y1=y1<0?0:(y1>Hs-1?Hs-1:y1);
  x0=x0<0?0:(x0>Ws-1?Ws-1:x0); x1=x1<0?0:(x1>Ws-1?Ws-1:x1);
  const float* p=s+((long)b*C+c)*Hs*Ws;
  float v00=p[y0*Ws+x0],v01=p[y0*Ws+x1],v10=p[y1*Ws+x0],v11=p[y1*Ws+x1];
  d[i]=v00*(1.f-fy)*(1.f-fx)+v01*(1.f-fy)*fx+v10*fy*(1.f-fx)+v11*fy*fx;
}
__global__ void k_sobel(const float* __restrict__ x,float* __restrict__ o,int B,int H,int W){
  long n=(long)B*H*W;
  long i=(long)blockIdx.x*blockDim.x+threadIdx.x; if(i>=n) return;
  int w=(int)(i%W); long r=i/W; int h=(int)(r%H); int b=(int)(r/H);
  const float* p=x+(long)b*H*W;
  auto at=[&](int hh,int ww)->float{
    return ((unsigned)hh<(unsigned)H && (unsigned)ww<(unsigned)W)?p[hh*W+ww]:0.f; };
  float a00=at(h-1,w-1),a01=at(h-1,w),a02=at(h-1,w+1);
  float a10=at(h,w-1),               a12=at(h,w+1);
  float a20=at(h+1,w-1),a21=at(h+1,w),a22=at(h+1,w+1);
  float gx=a00-a02+2.f*(a10-a12)+a20-a22;
  float gy=a00+2.f*a01+a02-a20-2.f*a21-a22;
  float g=sqrtf(gx*gx+gy*gy+1e-12f);
  o[i]=sigm_f(g)*p[h*W+w];
}
__global__ __launch_bounds__(256) void k_softmax(float* __restrict__ X,int cols){
  long row=blockIdx.x; float* r=X+row*(long)cols;
  __shared__ float red[256];
  int t=threadIdx.x;
  float m=-3.4e38f;
  for(int i=t;i<cols;i+=256) m=fmaxf(m,r[i]);
  red[t]=m; __syncthreads();
  for(int s=128;s>0;s>>=1){ if(t<s) red[t]=fmaxf(red[t],red[t+s]); __syncthreads(); }
  m=red[0]; __syncthreads();
  float sum=0.f;
  for(int i=t;i<cols;i+=256){ float e=__expf(r[i]-m); r[i]=e; sum+=e; }
  red[t]=sum; __syncthreads();
  for(int s=128;s>0;s>>=1){ if(t<s) red[t]+=red[t+s]; __syncthreads(); }
  float inv=1.f/red[0];
  for(int i=t;i<cols;i+=256) r[i]*=inv;
}

// ---------------------------------------------------------------- WMMA implicit-GEMM conv
// M = Cout, N = B*Ho*Wo, K = Cin*KH*KW (Kpad = round_up(K,32), weights pre-padded).
// A (weight) tile is TDM-fed; B (im2col patches) gathered by all threads.
#define TM 64
#define TN 128
#define TKK 32

__global__ __launch_bounds__(256)
void k_conv_wmma(const half_t* __restrict__ Wt,const half_t* __restrict__ X,
                 const float* __restrict__ bias,float* __restrict__ Y,
                 int Bn,int Cin,int H,int W,int Cout,int Ho,int Wo,
                 int KH,int KW,int stride,int pad,int dil,int Kpad)
{
  const long Npix=(long)Bn*Ho*Wo;
  const int hw=Ho*Wo;
  const int m0=blockIdx.y*TM;
  const long p0=(long)blockIdx.x*TN;
  const int t=threadIdx.x, wave=t>>5, lane=t&31;
  const int wm=wave&3, wn=wave>>2;
  const int K=Cin*KH*KW, khw=KH*KW;
  const int h=lane>>4, ml=lane&15;
  (void)wave;

  __shared__ __align__(16) half_t lA[2][TM*TKK];
  __shared__ __align__(16) half_t lB[2][TKK*TN];

  v8f acc[4]={};
  const int nK=Kpad/TKK;
  if(bias) __builtin_prefetch(bias+m0,0,2);        // global_prefetch_b8

  auto loadB=[&](int buf,int k0){
    int kk=t>>3, nc=(t&7)*16, gk=k0+kk;
    half_t vals[16];
    if(gk<K){
      int ci=gk/khw, r=gk%khw, kh=r/KW, kw=r%KW;
      #pragma unroll
      for(int i=0;i<16;i++){
        long p=p0+nc+i;
        half_t v=(half_t)0.f;
        if(p<Npix){
          int b=(int)(p/hw); int rem=(int)(p%hw);
          int ho=rem/Wo, wo=rem%Wo;
          int ih=ho*stride-pad+kh*dil, iw=wo*stride-pad+kw*dil;
          if((unsigned)ih<(unsigned)H && (unsigned)iw<(unsigned)W)
            v=X[(((size_t)b*Cin+ci)*H+ih)*W+iw];
        }
        vals[i]=v;
      }
    } else {
      #pragma unroll
      for(int i=0;i<16;i++) vals[i]=(half_t)0.f;
    }
    #pragma unroll
    for(int i=0;i<16;i++) lB[buf][kk*TN+nc+i]=vals[i];
  };
  auto mma=[&](int cur){
    v16h a;
    #pragma unroll
    for(int e=0;e<16;e++){ int ke=(e<8)?(e+8*h):(e+8+8*h); a[e]=lA[cur][(wm*16+ml)*TKK+ke]; }
    #pragma unroll
    for(int j=0;j<4;j++){
      v16h b;
      int nb=wn*64+j*16;
      #pragma unroll
      for(int e=0;e<16;e++) b[e]=lB[cur][((e<8)?(e+8*h):(e+8+8*h))*TN+nb+ml];
      acc[j]=__builtin_amdgcn_wmma_f32_16x16x32_f16(false,a,false,b,(short)0,acc[j],false,false);
    }
  };

#if HAVE_TDM
  auto tdmA=[&](int buf,int k0){
    unsigned rem1=(Cout>m0)?(unsigned)(Cout-m0):0u;
    tdm_load_2d(lds_off(&lA[buf][0]), Wt+(size_t)m0*Kpad+k0,
                TKK, TM, (unsigned)(Kpad-k0), rem1, (unsigned)Kpad);
  };
  if(t<32){ tdmA(0,0); if(nK>1) tdmA(1,TKK); }     // TDM ignores EXEC; one issue per wave0
  loadB(0,0);
  for(int ks=0;ks<nK;ks++){
    int cur=ks&1;
    if(t<32){                                      // leave the prefetched buffer in flight
      if(ks+1<nK) __builtin_amdgcn_s_wait_tensorcnt(1);
      else        __builtin_amdgcn_s_wait_tensorcnt(0);
    }
    __syncthreads();                               // lA[cur] (TDM) + lB[cur] visible to all waves
    if(ks+1<nK) loadB(cur^1,(ks+1)*TKK);
    mma(cur);
    __syncthreads();                               // everyone done with buffer `cur`
    if(t<32 && ks+2<nK) tdmA(cur,(ks+2)*TKK);      // recycle drained buffer
  }
#else
  auto loadA=[&](int buf,int k0){
    int row=t>>2, kc=(t&3)*8, gm=m0+row;
    if(gm<Cout){
      const half_t* s=Wt+(size_t)gm*Kpad+k0+kc;
      *(float4*)&lA[buf][row*TKK+kc]=*(const float4*)s;
    } else { float4 z={0.f,0.f,0.f,0.f}; *(float4*)&lA[buf][row*TKK+kc]=z; }
  };
  loadA(0,0); loadB(0,0);
  __syncthreads();
  for(int ks=0;ks<nK;ks++){
    int cur=ks&1;
    if(ks+1<nK){ loadA(cur^1,(ks+1)*TKK); loadB(cur^1,(ks+1)*TKK); }
    mma(cur);
    __syncthreads();
  }
#endif

  // C/D layout: VGPR r -> M = r + 8*(lane/16); N = lane%16
  #pragma unroll
  for(int j=0;j<4;j++){
    long p=p0+wn*64+j*16+ml;
    if(p<Npix){
      int b=(int)(p/hw); int rem=(int)(p%hw);
      #pragma unroll
      for(int r=0;r<8;r++){
        int co=m0+wm*16+8*h+r;
        if(co<Cout){
          float v=acc[j][r];
          if(bias) v+=bias[co];
          Y[((size_t)b*Cout+co)*hw+rem]=v;
        }
      }
    }
  }
}

// ---------------------------------------------------------------- batched WMMA GEMM (row-major)
// C[z] = alpha * A[z](MxKpad,f16) * B[z](KxN,f16) + biasN. Both tiles TDM-fed.
__global__ __launch_bounds__(256)
void k_gemm_wmma(const half_t* __restrict__ A,const half_t* __restrict__ Bm,
                 const float* __restrict__ biasN,float* __restrict__ C,
                 int M,int N,int K,int Kpad,long sA,long sB,long sC,float alpha)
{
  A+=(long)blockIdx.z*sA; Bm+=(long)blockIdx.z*sB; C+=(long)blockIdx.z*sC;
  const int m0=blockIdx.y*TM;
  const int n0=blockIdx.x*TN;
  const int t=threadIdx.x, wave=t>>5, lane=t&31;
  const int wm=wave&3, wn=wave>>2;
  const int h=lane>>4, ml=lane&15;
  (void)wave;
  __shared__ __align__(16) half_t lA[2][TM*TKK];
  __shared__ __align__(16) half_t lB[2][TKK*TN];
  v8f acc[4]={};
  const int nK=Kpad/TKK;

  auto mma=[&](int cur){
    v16h a;
    #pragma unroll
    for(int e=0;e<16;e++){ int ke=(e<8)?(e+8*h):(e+8+8*h); a[e]=lA[cur][(wm*16+ml)*TKK+ke]; }
    #pragma unroll
    for(int j=0;j<4;j++){
      v16h b;
      int nb=wn*64+j*16;
      #pragma unroll
      for(int e=0;e<16;e++) b[e]=lB[cur][((e<8)?(e+8*h):(e+8+8*h))*TN+nb+ml];
      acc[j]=__builtin_amdgcn_wmma_f32_16x16x32_f16(false,a,false,b,(short)0,acc[j],false,false);
    }
  };

#if HAVE_TDM
  auto tdmA=[&](int buf,int k0){
    unsigned rem1=(M>m0)?(unsigned)(M-m0):0u;
    tdm_load_2d(lds_off(&lA[buf][0]), A+(size_t)m0*Kpad+k0,
                TKK, TM, (unsigned)(Kpad-k0), rem1, (unsigned)Kpad);
  };
  auto tdmB=[&](int buf,int k0){
    unsigned rem0=(N>n0)?(unsigned)(N-n0):0u;     // OOB cols/rows zero-filled by TDM
    unsigned rem1=(K>k0)?(unsigned)(K-k0):0u;
    tdm_load_2d(lds_off(&lB[buf][0]), Bm+(size_t)k0*N+n0,
                TN, TKK, rem0, rem1, (unsigned)N);
  };
  if(t<32){ tdmA(0,0); tdmB(0,0); if(nK>1){ tdmA(1,TKK); tdmB(1,TKK); } }
  for(int ks=0;ks<nK;ks++){
    int cur=ks&1;
    if(t<32){
      if(ks+1<nK) __builtin_amdgcn_s_wait_tensorcnt(2);  // next buffer's 2 ops stay in flight
      else        __builtin_amdgcn_s_wait_tensorcnt(0);
    }
    __syncthreads();
    mma(cur);
    __syncthreads();
    if(t<32 && ks+2<nK){ tdmA(cur,(ks+2)*TKK); tdmB(cur,(ks+2)*TKK); }
  }
#else
  auto loadA=[&](int buf,int k0){
    int row=t>>2, kc=(t&3)*8, gm=m0+row;
    if(gm<M){
      const half_t* s=A+(size_t)gm*Kpad+k0+kc;
      *(float4*)&lA[buf][row*TKK+kc]=*(const float4*)s;
    } else { float4 z={0.f,0.f,0.f,0.f}; *(float4*)&lA[buf][row*TKK+kc]=z; }
  };
  auto loadB=[&](int buf,int k0){
    int kk=t>>3, nc=(t&7)*16, gk=k0+kk;
    #pragma unroll
    for(int i=0;i<16;i++){
      int col=n0+nc+i;
      half_t v=(half_t)0.f;
      if(gk<K && col<N) v=Bm[(size_t)gk*N+col];
      lB[buf][kk*TN+nc+i]=v;
    }
  };
  loadA(0,0); loadB(0,0);
  __syncthreads();
  for(int ks=0;ks<nK;ks++){
    int cur=ks&1;
    if(ks+1<nK){ loadA(cur^1,(ks+1)*TKK); loadB(cur^1,(ks+1)*TKK); }
    mma(cur);
    __syncthreads();
  }
#endif

  #pragma unroll
  for(int j=0;j<4;j++){
    int col=n0+wn*64+j*16+ml;
    if(col<N){
      #pragma unroll
      for(int r=0;r<8;r++){
        int row=m0+wm*16+8*h+r;
        if(row<M){
          float v=alpha*acc[j][r];
          if(biasN) v+=biasN[col];
          C[(size_t)row*N+col]=v;
        }
      }
    }
  }
}

// ---------------------------------------------------------------- host orchestration
struct Tns{ float* p; int C,H,W; };
struct ConvP{const float*b,*w;};
struct BnP{const float*b,*g;};
struct DownP{BnP bn1,bn2; ConvP c1,c11,c2,cf,pe;};
struct CccaeP{BnP bn1; ConvP c1x1,conv,pz1,pz2,pz3; BnP pzbn1,pzbn2,pzbn3;};
struct EdgeP{ConvP c11;};
struct UpP{BnP bn1,bn2; ConvP c1,c11,c2;};
struct SegP{BnP bn1; ConvP c1,c1x1f,c2,c3,d1,d2,d3,k,pe1,pe2,q,v;};

static inline dim3 g1(long n){ return dim3((unsigned)((n+255)/256)); }

extern "C" void kernel_launch(void* const* d_in, const int* in_sizes, int n_in,
                              void* d_out, int out_size, void* d_ws, size_t ws_size,
                              hipStream_t stream) {
  (void)in_sizes; (void)out_size; (void)ws_size;
  const int B=8; const int HW0=224*224;

  // ---- params: jax tree (sorted-key) flatten order ----
  int ci=0;
  auto nx=[&]{ return (const float*)d_in[ci++]; };
  auto rdC=[&]{ ConvP c; c.b=nx(); c.w=nx(); return c; };
  auto rdB=[&]{ BnP b; b.b=nx(); b.g=nx(); return b; };
  auto rdDown=[&]{ DownP d; d.bn1=rdB(); d.bn2=rdB(); d.c1=rdC(); d.c11=rdC();
                   d.c2=rdC(); d.cf=rdC(); d.pe=rdC(); return d; };
  auto rdCc=[&]{ CccaeP c; c.bn1=rdB(); c.c1x1=rdC(); c.conv=rdC(); c.pz1=rdC();
                 c.pz2=rdC(); c.pz3=rdC(); c.pzbn1=rdB(); c.pzbn2=rdB(); c.pzbn3=rdB(); return c; };
  auto rdE=[&]{ EdgeP e; e.c11=rdC(); return e; };
  auto rdUp=[&]{ UpP u; u.bn1=rdB(); u.bn2=rdB(); u.c1=rdC(); u.c11=rdC(); u.c2=rdC(); return u; };
  auto rdSeg=[&]{ SegP s; s.bn1=rdB(); s.c1=rdC(); s.c1x1f=rdC(); s.c2=rdC(); s.c3=rdC();
                  s.d1=rdC(); s.d2=rdC(); s.d3=rdC(); s.k=rdC(); s.pe1=rdC(); s.pe2=rdC();
                  s.q=rdC(); s.v=rdC(); return s; };

  DownP bottle=rdDown();
  CccaeP cc1=rdCc(), cc2=rdCc(), cc3=rdCc();
  DownP dn1=rdDown(),dn2=rdDown(),dn3=rdDown(),dn4=rdDown(),dn5=rdDown(),dn6=rdDown();
  EdgeP eg1=rdE(),eg2=rdE(),eg4=rdE(),eg5=rdE();
  SegP sg=rdSeg();
  UpP upP1=rdUp(),upP2=rdUp(),upP3=rdUp();
  const float* x=(const float*)d_in[n_in-1];

  // ---- stack allocator over d_ws (mark/restore recycles temporaries; stream order makes reuse safe)
  char* base=(char*)d_ws; size_t off=0;
  auto alloc=[&](size_t bytes)->void*{ size_t a=(off+255)&~(size_t)255; off=a+bytes; return base+a; };
  auto af=[&](size_t n)->float*{ return (float*)alloc(n*4); };
  auto ah=[&](size_t n)->half_t*{ return (half_t*)alloc(n*2); };

  auto toh=[&](const float* s,size_t n)->const half_t*{
    half_t* d=ah(n); k_f16<<<g1((long)n),256,0,stream>>>(s,d,(long)n); return d; };
  auto packW=[&](const float* s,int R,int K,int Kpad)->const half_t*{
    half_t* d=ah((size_t)R*Kpad); long n=(long)R*Kpad;
    k_packW<<<g1(n),256,0,stream>>>(s,d,R,K,Kpad); return d; };

  auto conv=[&](const float* X,int Cin,int H,int W,ConvP cp,int Cout,
                int KH,int KW,int st,int pad,int dil,float* Y,int Ho,int Wo){
    size_t mk=off;
    int K=Cin*KH*KW, Kpad=(K+31)&~31;
    const half_t* Wh=packW(cp.w,Cout,K,Kpad);
    const half_t* Xh=toh(X,(size_t)B*Cin*H*W);
    long Npix=(long)B*Ho*Wo;
    dim3 grid((unsigned)((Npix+TN-1)/TN),(unsigned)((Cout+TM-1)/TM),1);
    k_conv_wmma<<<grid,256,0,stream>>>(Wh,Xh,cp.b,Y,B,Cin,H,W,Cout,Ho,Wo,KH,KW,st,pad,dil,Kpad);
    off=mk;
  };
  auto bn_act=[&](float* X,int C,int HW,BnP p,int act){
    size_t mk=off;
    float2* mv=(float2*)alloc((size_t)C*sizeof(float2));
    k_bn_stats<<<dim3((unsigned)C),256,0,stream>>>(X,mv,B,C,HW);
    long n=(long)B*C*HW;
    k_bn_act<<<g1(n),256,0,stream>>>(X,mv,p.g,p.b,C,HW,n,act);
    off=mk;
  };
  auto catlist=[&](const float* const* ps,const int* cs,int k,int HW)->float*{
    int Ct=0; for(int i=0;i<k;i++) Ct+=cs[i];
    float* d=af((size_t)B*Ct*HW);
    int o2=0;
    for(int i=0;i<k;i++){
      k_chcpy<<<g1((long)B*cs[i]*HW),256,0,stream>>>(ps[i],d,B,cs[i],0,cs[i],HW,Ct,o2);
      o2+=cs[i];
    }
    return d;
  };
  auto cat2=[&](const float* A,int Ca,const float* Bp,int Cb,int HW)->float*{
    const float* ps[2]={A,Bp}; int cs[2]={Ca,Cb}; return catlist(ps,cs,2,HW); };

  auto down=[&](Tns xin,const DownP& P,int cout,int patch,bool pool,Tns& ypre)->Tns{
    int H=xin.H,W=xin.W,HW=H*W;
    float* y=af((size_t)B*cout*HW);
    float* pooled=pool?af((size_t)B*cout*(HW/4)):nullptr;
    size_t mk=off;
    int embed=xin.C*patch*patch;               // reshape back to (B, Cin, H, W) is a reinterpret
    float* vim=af((size_t)B*embed*(H/patch)*(W/patch));
    conv(xin.p,xin.C,H,W,P.pe,embed,patch,patch,patch,0,1,vim,H/patch,W/patch);
    float* res=af((size_t)B*cout*HW);
    conv(xin.p,xin.C,H,W,P.c11,cout,1,1,1,0,1,res,H,W);
    float* z1=af((size_t)B*cout*HW);
    conv(xin.p,xin.C,H,W,P.c1,cout,3,3,1,1,1,z1,H,W);
    bn_act(z1,cout,HW,P.bn1,1);
    float* z2=af((size_t)B*cout*HW);
    conv(z1,cout,H,W,P.c2,cout,3,3,1,1,1,z2,H,W);
    bn_act(z2,cout,HW,P.bn2,1);
    long n=(long)B*cout*HW;
    k_add<<<g1(n),256,0,stream>>>(z2,res,z2,n);
    float* cc=cat2(z2,cout,vim,xin.C,HW);
    conv(cc,cout+xin.C,H,W,P.cf,cout,3,3,1,1,1,y,H,W);
    k_add<<<g1(n),256,0,stream>>>(y,res,y,n);
    ypre={y,cout,H,W};
    Tns out=ypre;
    if(pool){ k_maxpool2<<<g1(n/4),256,0,stream>>>(y,pooled,B,cout,H,W); out={pooled,cout,H/2,W/2}; }
    off=mk;
    return out;
  };
  auto edge=[&](Tns y,const EdgeP& P)->float*{
    float* o=af((size_t)B*HW0);
    size_t mk=off;
    const float* r=y.p;
    if(y.H!=224){
      float* rr=af((size_t)B*y.C*HW0);
      k_resize<<<g1((long)B*y.C*HW0),256,0,stream>>>(y.p,rr,B,y.C,y.H,y.W,224,224);
      r=rr;
    }
    float* xc=af((size_t)B*HW0);
    conv(r,y.C,224,224,P.c11,1,1,1,1,0,1,xc,224,224);
    k_sobel<<<g1((long)B*HW0),256,0,stream>>>(xc,o,B,224,224);
    off=mk; return o;
  };
  auto cccae=[&](Tns e1,Tns e2,const CccaeP& P)->Tns{
    int H=e1.H,W=e1.W,HW=H*W, ch=e1.C, c=2*ch;
    float* out=af((size_t)B*c*HW);
    size_t mk=off;
    float* xx=cat2(e1.p,ch,e2.p,ch,HW);
    float* s1=af((size_t)B*ch*HW); conv(xx,c,H,W,P.pz1,ch,3,3,1,1,1,s1,H,W); bn_act(s1,ch,HW,P.pzbn1,2);
    float* s2=af((size_t)B*ch*HW); conv(xx,c,H,W,P.pz2,ch,3,3,1,2,2,s2,H,W); bn_act(s2,ch,HW,P.pzbn2,2);
    float* s3=af((size_t)B*ch*HW); conv(xx,c,H,W,P.pz3,ch,3,3,1,3,3,s3,H,W); bn_act(s3,ch,HW,P.pzbn3,2);
    const float* ps[3]={s1,s2,s3}; int cs[3]={ch,ch,ch};
    float* cc=catlist(ps,cs,3,HW);
    float* sa=af((size_t)B*ch*HW); conv(cc,3*ch,H,W,P.c1x1,ch,1,1,1,0,1,sa,H,W);
    long nh=(long)B*ch*HW;
    k_unary<<<g1(nh),256,0,stream>>>(sa,nh,3);
    float* m1=af((size_t)B*ch*HW); k_mul<<<g1(nh),256,0,stream>>>(e1.p,sa,m1,nh);
    float* m2=af((size_t)B*ch*HW); k_mul<<<g1(nh),256,0,stream>>>(e2.p,sa,m2,nh);
    const float* ps2[3]={xx,m1,m2}; int cs2[3]={c,ch,ch};
    float* oc=catlist(ps2,cs2,3,HW);
    conv(oc,2*c,H,W,P.conv,c,3,3,1,1,1,out,H,W);
    bn_act(out,c,HW,P.bn1,2);
    off=mk;
    return {out,c,H,W};
  };
  auto up=[&](Tns xin,Tns y,const UpP& P)->Tns{
    int H=y.H,W=y.W,HW=H*W, cout=y.C, cin=xin.C+y.C;
    float* out=af((size_t)B*cout*HW);
    size_t mk=off;
    const float* xr=xin.p;
    if(xin.H!=H){
      float* rr=af((size_t)B*xin.C*HW);
      k_resize<<<g1((long)B*xin.C*HW),256,0,stream>>>(xin.p,rr,B,xin.C,xin.H,xin.W,H,W);
      xr=rr;
    }
    float* cc=cat2(xr,xin.C,y.p,y.C,HW);
    float* res=af((size_t)B*cout*HW); conv(cc,cin,H,W,P.c11,cout,1,1,1,0,1,res,H,W);
    float* z=af((size_t)B*cout*HW);  conv(cc,cin,H,W,P.c1,cout,3,3,1,1,1,z,H,W); bn_act(z,cout,HW,P.bn1,1);
    conv(z,cout,H,W,P.c2,cout,3,3,1,1,1,out,H,W); bn_act(out,cout,HW,P.bn2,1);
    long n=(long)B*cout*HW;
    k_add<<<g1(n),256,0,stream>>>(out,res,out,n);
    off=mk;
    return {out,cout,H,W};
  };

  // ---- forward ----
  float* x1=af((size_t)B*2*HW0);           // [t1, t1ce]
  k_chcpy<<<g1((long)B*HW0),256,0,stream>>>(x,x1,B,4,1,1,HW0,2,0);
  k_chcpy<<<g1((long)B*HW0),256,0,stream>>>(x,x1,B,4,2,1,HW0,2,1);
  float* x2=af((size_t)B*2*HW0);           // [t2, flair]
  k_chcpy<<<g1((long)B*HW0),256,0,stream>>>(x,x2,B,4,3,1,HW0,2,0);
  k_chcpy<<<g1((long)B*HW0),256,0,stream>>>(x,x2,B,4,0,1,HW0,2,1);

  Tns y11,y12,y13,y21,y22,y23,dummy;
  Tns a1=down({x1,2,224,224},dn1,32,8,true,y11);
  Tns a2=down(a1,dn2,64,4,true,y12);
  Tns a3=down(a2,dn3,128,2,true,y13);
  float* e11=edge(y11,eg1); float* e12=edge(y12,eg2);
  Tns b1=down({x2,2,224,224},dn4,32,8,true,y21);
  Tns b2=down(b1,dn5,64,4,true,y22);
  Tns b3=down(b2,dn6,128,2,true,y23);
  float* e21=edge(y21,eg4); float* e22=edge(y22,eg5);
  const float* eps_[4]={e11,e12,e21,e22}; int ecs[4]={1,1,1,1};
  float* eC=catlist(eps_,ecs,4,HW0);               // (8,4,224,224)

  Tns yc1=cccae(y11,y21,cc1);
  Tns yc2=cccae(y12,y22,cc2);
  Tns yc3=cccae(y13,y23,cc3);

  float* xc=cat2(a3.p,128,b3.p,128,28*28);
  Tns xb=down({xc,256,28,28},bottle,512,1,false,dummy);

  Tns u3=up(xb,yc3,upP3);   // out1 (8,256,56,56)
  Tns u2=up(u3,yc2,upP2);   // out2 (8,128,112,112)
  Tns u1=up(u2,yc1,upP1);   // out3 (8,64,224,224)

  // ---- seg head ----
  {
    float* r1=af((size_t)B*u3.C*HW0);
    k_resize<<<g1((long)B*u3.C*HW0),256,0,stream>>>(u3.p,r1,B,u3.C,u3.H,u3.W,224,224);
    float* s1=af((size_t)B*12*HW0); conv(r1,u3.C,224,224,sg.c1,12,1,1,1,0,1,s1,224,224);
    float* r2=af((size_t)B*u2.C*HW0);
    k_resize<<<g1((long)B*u2.C*HW0),256,0,stream>>>(u2.p,r2,B,u2.C,u2.H,u2.W,224,224);
    float* s2=af((size_t)B*12*HW0); conv(r2,u2.C,224,224,sg.c2,12,1,1,1,0,1,s2,224,224);
    float* s3=af((size_t)B*12*HW0); conv(u1.p,u1.C,224,224,sg.c3,12,1,1,1,0,1,s3,224,224); // residual
    const float* ps[4]={s1,s2,s3,eC}; int cs[4]={12,12,12,4};
    float* cc=catlist(ps,cs,4,HW0);
    float* d1=af((size_t)B*12*HW0); conv(cc,40,224,224,sg.d1,12,3,3,1,1,1,d1,224,224);
    float* d2=af((size_t)B*12*HW0); conv(d1,12,224,224,sg.d2,12,3,3,1,2,2,d2,224,224);
    float* d3=af((size_t)B*12*HW0); conv(d2,12,224,224,sg.d3,12,3,3,1,3,3,d3,224,224);
    bn_act(d3,12,HW0,sg.bn1,1);

    const long TOK=784, D=768;
    float* tE=af((size_t)B*D*TOK); conv(d3,12,224,224,sg.pe1,768,8,8,8,0,1,tE,28,28);
    float* t =af((size_t)B*TOK*D);
    k_perm021<<<g1((long)B*D*TOK),256,0,stream>>>(tE,t,B,768,784);
    float* t3E=af((size_t)B*D*TOK); conv(s3,12,224,224,sg.pe2,768,8,8,8,0,1,t3E,28,28);
    float* t3=af((size_t)B*TOK*D);
    k_perm021<<<g1((long)B*D*TOK),256,0,stream>>>(t3E,t3,B,768,784);

    const half_t* tA =packW(t ,B*784,768,768);
    const half_t* t3A=packW(t3,B*784,768,768);
    const half_t* Wqh=toh(sg.q.w,768*768);
    const half_t* Wkh=toh(sg.k.w,768*768);
    const half_t* Wvh=toh(sg.v.w,768*768);
    float* q =af((size_t)B*TOK*D);
    float* kb=af((size_t)B*TOK*D);
    float* v =af((size_t)B*TOK*D);
    dim3 gl((768+TN-1)/TN,(784+TM-1)/TM,B);
    k_gemm_wmma<<<gl,256,0,stream>>>(tA ,Wqh,sg.q.b,q ,784,768,768,768,TOK*D,0,TOK*D,1.f);
    k_gemm_wmma<<<gl,256,0,stream>>>(tA ,Wkh,sg.k.b,kb,784,768,768,768,TOK*D,0,TOK*D,1.f);
    k_gemm_wmma<<<gl,256,0,stream>>>(t3A,Wvh,sg.v.b,v ,784,768,768,768,TOK*D,0,TOK*D,1.f);

    half_t* kT=ah((size_t)B*D*TOK);                      // k^T per batch, f16
    k_perm021h<<<g1((long)B*TOK*D),256,0,stream>>>(kb,kT,B,784,768);
    const half_t* qA=packW(q,B*784,768,768);
    float* sc=af((size_t)B*TOK*TOK);
    dim3 gs((784+TN-1)/TN,(784+TM-1)/TM,B);
    k_gemm_wmma<<<gs,256,0,stream>>>(qA,kT,nullptr,sc,784,784,768,768,TOK*D,D*TOK,TOK*TOK,
                                     0.03608439182435161f /* 1/sqrt(768) */);
    k_softmax<<<dim3((unsigned)(B*784)),256,0,stream>>>(sc,784);
    const half_t* aA=packW(sc,B*784,784,800);
    const half_t* vh=toh(v,(size_t)B*TOK*D);
    float* o=af((size_t)B*TOK*D);
    dim3 go((768+TN-1)/TN,(784+TM-1)/TM,B);
    k_gemm_wmma<<<go,256,0,stream>>>(aA,vh,nullptr,o,784,768,784,800,TOK*800,TOK*D,TOK*D,1.f);
    float* om=af((size_t)B*D*TOK);                       // (B,768,784) == (8,12,224,224)
    k_perm021<<<g1((long)B*TOK*D),256,0,stream>>>(o,om,B,784,768);

    const float* fps[2]={s3,om}; int fcs[2]={12,12};
    float* fc=catlist(fps,fcs,2,HW0);
    conv(fc,24,224,224,sg.c1x1f,3,1,1,1,0,1,(float*)d_out,224,224);
  }
}
